// SingleMotionPredictor_80616536146334
// MI455X (gfx1250) — matrix operand order repaired and checked by reading, not compiled, not run on hardware
//
#include <hip/hip_runtime.h>
#include <hip/hip_bf16.h>
#include <stdint.h>

// ---------------- problem constants ----------------
#define T_STEPS 64
#define BATCH   512
#define FEAT    72
#define HID     1024
#define OUTL    32
#define GATES   4096          // 4*HID
#define K0      1120          // FEAT + HID padded to 35*32
#define K1      2048          // HID + HID
#define NLIN    128           // FEAT padded to 8*16 for the linear GEMM

#define LDSS    40            // LDS row stride in bf16 elements (80 B = 20 banks)

typedef __attribute__((ext_vector_type(16))) __bf16 v16bf;
typedef __attribute__((ext_vector_type(8)))  __bf16 v8bf;
typedef __attribute__((ext_vector_type(8)))  float  v8f;
typedef __attribute__((ext_vector_type(4)))  unsigned int v4u;   // 16-byte chunk

__device__ __forceinline__ unsigned short f2bf(float f) {
    union { float f; unsigned int u; } v; v.f = f;
    unsigned int u = v.u;
    unsigned int r = u + 0x7FFFu + ((u >> 16) & 1u);   // round-to-nearest-even
    return (unsigned short)(r >> 16);
}

__device__ __forceinline__ float sigmoidf_fast(float x) {
    return 1.0f / (1.0f + __expf(-x));
}

// ---------------------------------------------------------------------------
// bf16 WMMA GEMM:  C[M x N] = A[M x K] * B[N x K]^T   (both row-major, fp32 C)
// M = BATCH = 512.  K multiple of 32, N multiple of 128.
// Block: 256 threads (8 waves).  Wave tile 64(M) x 32(N) -> block 128 x 128.
// K-tiles of 32 are double-buffered in LDS (global -> regs -> ds_store), and
// WMMA fragments are served from LDS in contiguous k order (k = 16*half + s, A and B alike).
// LDS rows padded to 80 B so each 16-lane half of a b128 fragment read hits
// distinct bank quads (optimal 2-cycle service for 512 B/wave).
// ---------------------------------------------------------------------------
__global__ __launch_bounds__(256)
void gemm_bf16_nt(const unsigned short* __restrict__ A, int lda,
                  const unsigned short* __restrict__ B, int ldb,
                  float* __restrict__ C, int ldc, int K)
{
    __shared__ unsigned short As[2][128 * LDSS];   // 2 x 10 KB
    __shared__ unsigned short Bs[2][128 * LDSS];   // 2 x 10 KB  (40 KB total)

    const int tid  = threadIdx.x;
    const int lane = tid & 31;
    const int wave = tid >> 5;
    const int wm   = wave >> 2;                 // 0..1
    const int wn   = wave & 3;                  // 0..3
    const int l    = lane & 15;
    const bool hi  = lane >= 16;

    // staging assignment: thread -> (row, 16-element column half) of the K-tile
    const int srow = tid >> 1;                  // 0..127
    const int scol = (tid & 1) << 4;            // 0 or 16

    const unsigned short* aSrc = A + (size_t)(blockIdx.y * 128 + srow) * lda + scol;
    const unsigned short* bSrc = B + (size_t)(blockIdx.x * 128 + srow) * ldb + scol;

    v8f acc[4][2];
#pragma unroll
    for (int i = 0; i < 4; ++i)
#pragma unroll
        for (int j = 0; j < 2; ++j)
            acc[i][j] = (v8f){0.f,0.f,0.f,0.f,0.f,0.f,0.f,0.f};

    // ---- prologue: stage K-tile 0 into buffer 0
    v4u pa0 = *(const v4u*)(aSrc);
    v4u pa1 = *(const v4u*)(aSrc + 8);
    v4u pb0 = *(const v4u*)(bSrc);
    v4u pb1 = *(const v4u*)(bSrc + 8);
    {
        unsigned short* d = &As[0][srow * LDSS + scol];
        *(v4u*)d = pa0; *(v4u*)(d + 8) = pa1;
        d = &Bs[0][srow * LDSS + scol];
        *(v4u*)d = pb0; *(v4u*)(d + 8) = pb1;
    }
    __syncthreads();

    int buf = 0;
    for (int kb = 0; kb < K; kb += 32) {
        // ---- prefetch next K-tile into registers (dummy re-read on last iter)
        const int kn = (kb + 32 < K) ? (kb + 32) : 0;
        pa0 = *(const v4u*)(aSrc + kn);
        pa1 = *(const v4u*)(aSrc + kn + 8);
        pb0 = *(const v4u*)(bSrc + kn);
        pb1 = *(const v4u*)(bSrc + kn + 8);

        // ---- B fragments from LDS: lane n (per half-wave) = row n of W
        v16bf bfr[2];
#pragma unroll
        for (int j = 0; j < 2; ++j)
            bfr[j] = *(const v16bf*)
                &Bs[buf][(wn * 32 + j * 16 + l) * LDSS + (hi ? 16 : 0)];

        // ---- A fragments from LDS + 8 WMMAs
#pragma unroll
        for (int i = 0; i < 4; ++i) {
            const unsigned short* ap =
                &As[buf][(wm * 64 + i * 16 + l) * LDSS + (hi ? 16 : 0)];
            v8bf x0 = *(const v8bf*)ap;          // K 0..7   (16..23 for hi lanes)
            v8bf x1 = *(const v8bf*)(ap + 8);    // K 8..15  (24..31 for hi lanes)
            v16bf afr = __builtin_shufflevector(x0, x1,
                0,1,2,3,4,5,6,7,8,9,10,11,12,13,14,15);
#pragma unroll
            for (int j = 0; j < 2; ++j) {
                acc[i][j] = __builtin_amdgcn_wmma_f32_16x16x32_bf16(
                    /*neg_a=*/false, afr, /*neg_b=*/false, bfr[j],
                    /*c_mod=*/(short)0, acc[i][j],
                    /*reuse_a=*/false, /*reuse_b=*/false);
            }
        }

        // ---- stage next tile into the other buffer; one barrier per K-iter
        {
            unsigned short* d = &As[buf ^ 1][srow * LDSS + scol];
            *(v4u*)d = pa0; *(v4u*)(d + 8) = pa1;
            d = &Bs[buf ^ 1][srow * LDSS + scol];
            *(v4u*)d = pb0; *(v4u*)(d + 8) = pb1;
        }
        __syncthreads();
        buf ^= 1;
    }

    // ---- store C: VGPR r -> row m0+i*16+(hi?8:0)+r, col n0+j*16+l
    const int m0 = blockIdx.y * 128 + wm * 64;
    const int n0 = blockIdx.x * 128 + wn * 32;
    const int cmBase = m0 + (hi ? 8 : 0);
#pragma unroll
    for (int i = 0; i < 4; ++i)
#pragma unroll
        for (int j = 0; j < 2; ++j) {
            float* cp = C + (size_t)(cmBase + i * 16) * ldc + n0 + j * 16 + l;
#pragma unroll
            for (int r = 0; r < 8; ++r)
                cp[(size_t)r * ldc] = acc[i][j][r];
        }
}

// ---------------------------------------------------------------------------
// Weight / bias repack kernels (fp32 -> bf16, concat + zero pad)
// ---------------------------------------------------------------------------
__global__ void pack_w0(const float* __restrict__ Wih,   // (4096,72)
                        const float* __restrict__ Whh,   // (4096,1024)
                        unsigned short* __restrict__ W)  // (4096,K0)
{
    int idx = blockIdx.x * blockDim.x + threadIdx.x;
    if (idx >= GATES * K0) return;
    int n = idx / K0, k = idx % K0;
    float v = 0.f;
    if (k < FEAT)            v = Wih[(size_t)n * FEAT + k];
    else if (k < FEAT + HID) v = Whh[(size_t)n * HID + (k - FEAT)];
    W[idx] = f2bf(v);
}

__global__ void pack_w1(const float* __restrict__ Wih,   // (4096,1024)
                        const float* __restrict__ Whh,   // (4096,1024)
                        unsigned short* __restrict__ W)  // (4096,K1)
{
    int idx = blockIdx.x * blockDim.x + threadIdx.x;
    if (idx >= GATES * K1) return;
    int n = idx / K1, k = idx % K1;
    float v = (k < HID) ? Wih[(size_t)n * HID + k]
                        : Whh[(size_t)n * HID + (k - HID)];
    W[idx] = f2bf(v);
}

__global__ void pack_wlin(const float* __restrict__ Wl,  // (72,1024)
                          unsigned short* __restrict__ W) // (128,1024)
{
    int idx = blockIdx.x * blockDim.x + threadIdx.x;
    if (idx >= NLIN * HID) return;
    int n = idx / HID, k = idx % HID;
    float v = (n < FEAT) ? Wl[(size_t)n * HID + k] : 0.f;
    W[idx] = f2bf(v);
}

__global__ void bias_combine(const float* __restrict__ a,
                             const float* __restrict__ b,
                             float* __restrict__ o)
{
    int idx = blockIdx.x * blockDim.x + threadIdx.x;
    if (idx < GATES) o[idx] = a[idx] + b[idx];
}

__global__ void zero_state(unsigned short* h0, unsigned short* h1,
                           float* c0, float* c1)
{
    int idx = blockIdx.x * blockDim.x + threadIdx.x;
    if (idx >= BATCH * HID) return;
    h0[idx] = 0; h1[idx] = 0; c0[idx] = 0.f; c1[idx] = 0.f;
}

__global__ void init_fb(const float* __restrict__ batch,
                        const int* __restrict__ lengths,
                        const float* __restrict__ mean,
                        const float* __restrict__ stdv,
                        float* __restrict__ fb)
{
    int idx = blockIdx.x * blockDim.x + threadIdx.x;
    if (idx >= BATCH * FEAT) return;
    int b = idx / FEAT, j = idx % FEAT;
    int t = lengths[b] - 1;
    float x = batch[((size_t)t * BATCH + b) * FEAT + j];
    fb[idx] = (x - mean[j]) / stdv[j];
}

// Build packed cell input X = [x(72) | h(1024) | pad] in bf16 (row len K0)
__global__ void pack_x0(const float* __restrict__ xsrc,   // (B,F) raw or fb
                        const float* __restrict__ mean,
                        const float* __restrict__ stdv,
                        int do_norm,
                        const unsigned short* __restrict__ h,
                        unsigned short* __restrict__ X)
{
    int idx = blockIdx.x * blockDim.x + threadIdx.x;
    if (idx >= BATCH * K0) return;
    int b = idx / K0, j = idx % K0;
    unsigned short v;
    if (j < FEAT) {
        float x = xsrc[(size_t)b * FEAT + j];
        if (do_norm) x = (x - mean[j]) / stdv[j];
        v = f2bf(x);
    } else if (j < FEAT + HID) {
        v = h[(size_t)b * HID + (j - FEAT)];
    } else v = 0;
    X[idx] = v;
}

// Copy a hidden state (bf16) into a slice of a packed input matrix
__global__ void pack_h_to_x(const unsigned short* __restrict__ h,
                            unsigned short* __restrict__ X,
                            int xstride, int xoff)
{
    int idx = blockIdx.x * blockDim.x + threadIdx.x;
    if (idx >= BATCH * HID) return;
    int b = idx / HID, j = idx % HID;
    X[(size_t)b * xstride + xoff + j] = h[idx];
}

// Fused LSTM elementwise update.  Writes unmasked hn into xnext (layer l+1
// input), and masked h/c state updates.
__global__ void lstm_update(const float* __restrict__ gates,  // (B,4096)
                            const float* __restrict__ bias,   // (4096)
                            float* __restrict__ c,            // (B,H)
                            unsigned short* __restrict__ h,   // (B,H) bf16
                            unsigned short* __restrict__ xnext,
                            int xstride, int xoff,
                            const int* __restrict__ lengths,
                            int t, int use_mask)
{
    int idx = blockIdx.x * blockDim.x + threadIdx.x;
    if (idx >= BATCH * HID) return;
    int b = idx / HID, j = idx % HID;
    const float* g = gates + (size_t)b * GATES;
    float gi = g[j]           + bias[j];
    float gf = g[HID + j]     + bias[HID + j];
    float gg = g[2 * HID + j] + bias[2 * HID + j];
    float go = g[3 * HID + j] + bias[3 * HID + j];
    float cn = sigmoidf_fast(gf) * c[idx] + sigmoidf_fast(gi) * tanhf(gg);
    float hn = sigmoidf_fast(go) * tanhf(cn);
    unsigned short hnb = f2bf(hn);
    if (xnext) xnext[(size_t)b * xstride + xoff + j] = hnb;  // unmasked feed-forward
    bool m = use_mask ? (t < lengths[b] - 1) : true;
    if (m) { c[idx] = cn; h[idx] = hnb; }
}

// Decoder feedback + denormalized output
__global__ void fb_out(const float* __restrict__ tmp,   // (B,NLIN)
                       const float* __restrict__ blin,  // (72)
                       float* __restrict__ fb,          // (B,72)
                       float* __restrict__ out,         // (B,72) at step s
                       const float* __restrict__ mean,
                       const float* __restrict__ stdv)
{
    int idx = blockIdx.x * blockDim.x + threadIdx.x;
    if (idx >= BATCH * FEAT) return;
    int b = idx / FEAT, j = idx % FEAT;
    float v = fb[idx] + tmp[(size_t)b * NLIN + j] + blin[j];
    fb[idx]  = v;
    out[idx] = v * stdv[j] + mean[j];
}

// ---------------------------------------------------------------------------
extern "C" void kernel_launch(void* const* d_in, const int* in_sizes, int n_in,
                              void* d_out, int out_size, void* d_ws, size_t ws_size,
                              hipStream_t stream)
{
    (void)in_sizes; (void)n_in; (void)out_size; (void)ws_size;

    const float* batch = (const float*)d_in[0];
    const int*   lens  = (const int*)d_in[1];
    const float* mean  = (const float*)d_in[3];
    const float* stdv  = (const float*)d_in[4];
    const float* Wih0  = (const float*)d_in[5];
    const float* Whh0  = (const float*)d_in[6];
    const float* bih0  = (const float*)d_in[7];
    const float* bhh0  = (const float*)d_in[8];
    const float* Wih1  = (const float*)d_in[9];
    const float* Whh1  = (const float*)d_in[10];
    const float* bih1  = (const float*)d_in[11];
    const float* bhh1  = (const float*)d_in[12];
    const float* Wlin  = (const float*)d_in[13];
    const float* blin  = (const float*)d_in[14];
    float* out = (float*)d_out;

    // -------- workspace layout (all sizes multiple of 256 bytes) --------
    char* ws = (char*)d_ws;
    auto take = [&](size_t bytes) { char* p = ws; ws += bytes; return p; };
    unsigned short* W0    = (unsigned short*)take((size_t)GATES * K0 * 2);  // 9.2 MB
    unsigned short* W1    = (unsigned short*)take((size_t)GATES * K1 * 2);  // 16.8 MB
    unsigned short* WL    = (unsigned short*)take((size_t)NLIN * HID * 2);  // 0.25 MB
    float*          bias0 = (float*)take((size_t)GATES * 4);
    float*          bias1 = (float*)take((size_t)GATES * 4);
    unsigned short* h0    = (unsigned short*)take((size_t)BATCH * HID * 2);
    unsigned short* h1    = (unsigned short*)take((size_t)BATCH * HID * 2);
    float*          c0    = (float*)take((size_t)BATCH * HID * 4);
    float*          c1    = (float*)take((size_t)BATCH * HID * 4);
    unsigned short* X0    = (unsigned short*)take((size_t)BATCH * K0 * 2);
    unsigned short* X1    = (unsigned short*)take((size_t)BATCH * K1 * 2);
    float*          gates = (float*)take((size_t)BATCH * GATES * 4);        // 8 MB
    float*          tmpL  = (float*)take((size_t)BATCH * NLIN * 4);
    float*          fb    = (float*)take((size_t)BATCH * FEAT * 4);

    const int TB = 256;
    auto blk = [](size_t n, int tb) { return (unsigned)((n + tb - 1) / tb); };

    // -------- one-time (per-launch) weight repack --------
    pack_w0<<<blk((size_t)GATES * K0, TB), TB, 0, stream>>>(Wih0, Whh0, W0);
    pack_w1<<<blk((size_t)GATES * K1, TB), TB, 0, stream>>>(Wih1, Whh1, W1);
    pack_wlin<<<blk((size_t)NLIN * HID, TB), TB, 0, stream>>>(Wlin, WL);
    bias_combine<<<blk(GATES, TB), TB, 0, stream>>>(bih0, bhh0, bias0);
    bias_combine<<<blk(GATES, TB), TB, 0, stream>>>(bih1, bhh1, bias1);
    zero_state<<<blk((size_t)BATCH * HID, TB), TB, 0, stream>>>(h0, h1, c0, c1);
    init_fb<<<blk((size_t)BATCH * FEAT, TB), TB, 0, stream>>>(batch, lens, mean, stdv, fb);

    const dim3 gGemm(GATES / 128, BATCH / 128);   // 32 x 4
    const dim3 gLin(NLIN / 128, BATCH / 128);     // 1 x 4
    const unsigned nBH  = blk((size_t)BATCH * HID, TB);
    const unsigned nBK0 = blk((size_t)BATCH * K0, TB);
    const unsigned nBF  = blk((size_t)BATCH * FEAT, TB);

    // -------- encoder: 63 masked steps --------
    for (int t = 0; t < T_STEPS - 1; ++t) {
        pack_x0<<<nBK0, TB, 0, stream>>>(batch + (size_t)t * BATCH * FEAT,
                                         mean, stdv, 1, h0, X0);
        gemm_bf16_nt<<<gGemm, 256, 0, stream>>>(X0, K0, W0, K0, gates, GATES, K0);
        lstm_update<<<nBH, TB, 0, stream>>>(gates, bias0, c0, h0,
                                            X1, K1, 0, lens, t, 1);
        pack_h_to_x<<<nBH, TB, 0, stream>>>(h1, X1, K1, HID);
        gemm_bf16_nt<<<gGemm, 256, 0, stream>>>(X1, K1, W1, K1, gates, GATES, K1);
        lstm_update<<<nBH, TB, 0, stream>>>(gates, bias1, c1, h1,
                                            nullptr, 0, 0, lens, t, 1);
    }

    // -------- decoder: 32 steps with linear feedback --------
    for (int s = 0; s < OUTL; ++s) {
        pack_x0<<<nBK0, TB, 0, stream>>>(fb, mean, stdv, 0, h0, X0);
        gemm_bf16_nt<<<gGemm, 256, 0, stream>>>(X0, K0, W0, K0, gates, GATES, K0);
        lstm_update<<<nBH, TB, 0, stream>>>(gates, bias0, c0, h0,
                                            X1, K1, 0, lens, 0, 0);
        pack_h_to_x<<<nBH, TB, 0, stream>>>(h1, X1, K1, HID);
        gemm_bf16_nt<<<gGemm, 256, 0, stream>>>(X1, K1, W1, K1, gates, GATES, K1);
        lstm_update<<<nBH, TB, 0, stream>>>(gates, bias1, c1, h1,
                                            nullptr, 0, 0, lens, 0, 0);
        // top = new h1 (bf16) ; tmpL = top @ Wlin^T (N padded to 128)
        gemm_bf16_nt<<<gLin, 256, 0, stream>>>(h1, HID, WL, HID, tmpL, NLIN, HID);
        fb_out<<<nBF, TB, 0, stream>>>(tmpL, blin, fb,
                                       out + (size_t)s * BATCH * FEAT, mean, stdv);
    }
}
